// Transformer_80659485819421
// MI455X (gfx1250) — compile-verified, matrix operand order checked by reading, not run on hardware
//
#include <hip/hip_runtime.h>
#include <math.h>

typedef unsigned short u16;
typedef __attribute__((ext_vector_type(16))) __bf16 bf16x16;
typedef __attribute__((ext_vector_type(8)))  float  floatx8;
typedef __attribute__((ext_vector_type(4)))  unsigned int u32x4;
typedef __attribute__((ext_vector_type(4)))  int i32x4;
typedef __attribute__((ext_vector_type(8)))  int i32x8;

union FragA { bf16x16 v; u16 u[16]; uint4 u4[2]; };
union FragC { floatx8 v; float f[8]; };

__device__ __forceinline__ u16 f2bf(float f) {
  unsigned u = __float_as_uint(f);
  u += 0x7fffu + ((u >> 16) & 1u);          // round-to-nearest-even
  return (u16)(u >> 16);
}

__device__ __forceinline__ floatx8 wmma_bf16(bf16x16 a, bf16x16 b, floatx8 c) {
  // D = A(16x32 bf16) * B(32x16 bf16) + C(16x16 f32)
  return __builtin_amdgcn_wmma_f32_16x16x32_bf16(false, a, false, b, (short)0, c,
                                                 false, false);
}

// ---------------------------------------------------------------------------
// Tensor Data Mover: 2D tile load (global -> LDS), 2-byte elements.
// Tile: tile_dim0 = 32 elements contiguous (K), tile_dim1 = 128 rows,
// row stride = K elements. LDS padding: 4 DWORDs after every 16 DWORDs
// (pad_interval=3, pad_amount=3) -> LDS pitch = 40 u16 per row, so the
// consumers' ds_load_b128 fragment reads stay 16B-aligned and conflict-free.
// D# bitfields per CDNA5 ISA.
// ---------------------------------------------------------------------------
__device__ __forceinline__ void tdm_load_2d(const u16* gbase, unsigned lds_off,
                                            unsigned tensor_d0, unsigned tensor_d1,
                                            unsigned stride_d0)
{
  unsigned long long ga = (unsigned long long)gbase;
  u32x4 g0;
  g0[0] = 1u;                                           // count=1, user mode
  g0[1] = lds_off;                                      // lds_addr (bytes)
  g0[2] = (unsigned)(ga & 0xffffffffu);                 // global_addr[31:0]
  g0[3] = (unsigned)((ga >> 32) & 0x01ffffffu)          // global_addr[56:32]
        | (2u << 30);                                   // type = 2 (image)
  i32x8 g1;
  g1[0] = (int)((1u << 16)        // data_size = 1 -> 2 bytes
              | (1u << 20)        // pad_enable
              | (3u << 22)        // pad_interval: 16 DWORDs
              | (3u << 25));      // pad_amount:   4 DWORDs
  g1[1] = (int)((tensor_d0 & 0xffffu) << 16);                           // dim0 lo16
  g1[2] = (int)(((tensor_d0 >> 16) & 0xffffu) | ((tensor_d1 & 0xffffu) << 16));
  g1[3] = (int)(((tensor_d1 >> 16) & 0xffffu) | (32u << 16));           // tile_dim0=32
  g1[4] = (int)128;                                                     // tile_dim1=128
  g1[5] = (int)stride_d0;                                               // dim0 stride lo
  g1[6] = 0;
  g1[7] = 0;
  i32x4 z4 = {0, 0, 0, 0};
#if __clang_major__ >= 23
  i32x8 z8 = {0, 0, 0, 0, 0, 0, 0, 0};
  __builtin_amdgcn_tensor_load_to_lds(g0, g1, z4, z4, z8, 0);
#else
  __builtin_amdgcn_tensor_load_to_lds(g0, g1, z4, z4, 0);
#endif
}

// ---------------------------------------------------------------------------
// TDM-fed bf16 GEMM: C[M,N] = A[M,K] * BT[N,K]^T (+bias),
// optional fused residual+layerscale: C = res + ls * (acc + bias).
// Block: 256 threads = 8 waves; block tile 128x128, K-step 32, LDS
// double-buffered. Wave (wm,wn) owns 32x64 -> 8 WMMA accumulators, fed by
// 12 ds_load_b128 per K-step. Tiles move global->LDS via tensor_load_to_lds
// issued by wave 0, synchronized with s_wait_tensorcnt + barrier.
// Dynamic LDS layout (u16): A0[0,5120) A1[5120,10240) B0[10240,15360)
// B1[15360,20480)  => 40960 bytes.
// ---------------------------------------------------------------------------
__global__ __launch_bounds__(256) void gemm_bf16_kernel(
    const u16* __restrict__ A, const u16* __restrict__ BT,
    const float* __restrict__ bias, const float* __restrict__ res,
    const float* __restrict__ ls, float* __restrict__ C,
    int M, int N, int K)
{
  extern __shared__ u16 smem[];
  const int tid  = threadIdx.x;
  const int wv   = tid >> 5;
  const int lane = tid & 31;
  const int wm   = wv >> 1;                 // 0..3 : 32-row strip
  const int wn   = wv & 1;                  // 0..1 : 64-col strip
  const int h    = lane >> 4;
  const int l16  = lane & 15;
  const int bm   = blockIdx.y << 7;
  const int bn   = blockIdx.x << 7;

  FragC acc[2][4];
#pragma unroll
  for (int i = 0; i < 2; ++i)
#pragma unroll
    for (int j = 0; j < 4; ++j)
#pragma unroll
      for (int r = 0; r < 8; ++r) acc[i][j].f[r] = 0.f;

  const u16* Ag = A  + (size_t)bm * K;      // 128-row A panel, row stride K
  const u16* Bg = BT + (size_t)bn * K;      // 128-row B^T panel, row stride K

  if (wv == 0) {                            // prologue: fill buffer 0
    tdm_load_2d(Ag, 0u,     (unsigned)K, (unsigned)M, (unsigned)K);
    tdm_load_2d(Bg, 20480u, (unsigned)K, (unsigned)N, (unsigned)K);
  }

  int cur = 0;
  for (int k0 = 0; k0 < K; k0 += 32) {
    const int  nxt  = cur ^ 1;
    const bool more = (k0 + 32) < K;
    if (wv == 0) {
      if (more) {
        tdm_load_2d(Ag + k0 + 32, (unsigned)(nxt * 10240),
                    (unsigned)K, (unsigned)M, (unsigned)K);
        tdm_load_2d(Bg + k0 + 32, (unsigned)(20480 + nxt * 10240),
                    (unsigned)K, (unsigned)N, (unsigned)K);
        __builtin_amdgcn_s_wait_tensorcnt((short)2);  // current pair landed
      } else {
        __builtin_amdgcn_s_wait_tensorcnt((short)0);
      }
    }
    __syncthreads();                        // tile `cur` visible to all waves

    const u16* Ab = smem + cur * 5120;
    const u16* Bb = smem + 10240 + cur * 5120;
    FragA af[2], bfr[4];
#pragma unroll
    for (int i = 0; i < 2; ++i) {
      const u16* p = Ab + (wm * 32 + i * 16 + l16) * 40 + h * 8;
      af[i].u4[0] = *(const uint4*)(p);
      af[i].u4[1] = *(const uint4*)(p + 16);
    }
#pragma unroll
    for (int j = 0; j < 4; ++j) {
      const u16* p = Bb + (wn * 64 + j * 16 + l16) * 40 + h * 8;
      bfr[j].u4[0] = *(const uint4*)(p);
      bfr[j].u4[1] = *(const uint4*)(p + 16);
    }
#pragma unroll
    for (int i = 0; i < 2; ++i)
#pragma unroll
      for (int j = 0; j < 4; ++j)
        acc[i][j].v = wmma_bf16(af[i].v, bfr[j].v, acc[i][j].v);

    __syncthreads();                        // done reading before TDM overwrites
    cur = nxt;
  }

#pragma unroll
  for (int i = 0; i < 2; ++i)
#pragma unroll
    for (int j = 0; j < 4; ++j) {
      const int nb = bn + wn * 64 + j * 16 + l16;
      const float bvv = bias ? bias[nb] : 0.f;
      const float lv  = ls ? ls[nb] : 0.f;
#pragma unroll
      for (int r = 0; r < 8; ++r) {
        const int mr = bm + wm * 32 + i * 16 + h * 8 + r;   // C: M = r + 8h
        const size_t idx = (size_t)mr * N + nb;
        const float v = acc[i][j].f[r] + bvv;
        C[idx] = res ? (res[idx] + lv * v) : v;
      }
    }
}

// ---------------------------------------------------------------------------
// Flash attention (wave32): one wave per (head, 16-row Q tile), 32-key chunks.
// Qh: [H][nq][64] bf16 (pre-scaled), Kh: [H][nk][64] bf16,
// Vt: [H][64][nk] bf16 (dim-major so V fragments load as b128).
// O:  [nq][H*64] bf16 merged-heads.
// ---------------------------------------------------------------------------
__global__ __launch_bounds__(256) void attn_kernel(
    const u16* __restrict__ Qh, const u16* __restrict__ Kh,
    const u16* __restrict__ Vt, u16* __restrict__ O,
    int nq, int nk, int causal)
{
  __shared__ u16 Pl[4][16 * 40];   // per-wave P-tile staging (C->A relayout)
  const int tid = threadIdx.x, wv = tid >> 5, lane = tid & 31;
  const int h = lane >> 4, l16 = lane & 15;
  const int head = blockIdx.y;
  const int qt = (blockIdx.x << 6) + wv * 16;

  const u16* Q  = Qh + (size_t)head * nq * 64;
  const u16* Kp = Kh + (size_t)head * nk * 64;
  const u16* Vp = Vt + (size_t)head * 64 * nk;

  FragA aq[2];
#pragma unroll
  for (int s = 0; s < 2; ++s) {
    const u16* qrow = Q + (size_t)(qt + l16) * 64 + s * 32 + h * 8;
    aq[s].u4[0] = *(const uint4*)(qrow);
    aq[s].u4[1] = *(const uint4*)(qrow + 16);
  }

  FragC o[4];
#pragma unroll
  for (int n = 0; n < 4; ++n)
#pragma unroll
    for (int r = 0; r < 8; ++r) o[n].f[r] = 0.f;
  float mrow[8], lrow[8];
#pragma unroll
  for (int r = 0; r < 8; ++r) { mrow[r] = -1e30f; lrow[r] = 0.f; }

  const int kend = causal ? (qt + 16) : nk;
  for (int kc = 0; kc < kend; kc += 32) {
    FragC s0, s1;
#pragma unroll
    for (int r = 0; r < 8; ++r) { s0.f[r] = 0.f; s1.f[r] = 0.f; }
#pragma unroll
    for (int s = 0; s < 2; ++s) {             // head-dim in 2x32
      FragA bk0, bk1;
      const u16* kr0 = Kp + (size_t)(kc + l16) * 64 + s * 32 + h * 8;
      bk0.u4[0] = *(const uint4*)(kr0);
      bk0.u4[1] = *(const uint4*)(kr0 + 16);
      const u16* kr1 = Kp + (size_t)(kc + 16 + l16) * 64 + s * 32 + h * 8;
      bk1.u4[0] = *(const uint4*)(kr1);
      bk1.u4[1] = *(const uint4*)(kr1 + 16);
      s0.v = wmma_bf16(aq[s].v, bk0.v, s0.v);
      s1.v = wmma_bf16(aq[s].v, bk1.v, s1.v);
    }
    // online softmax; row stats live across the 16 lanes of each half-wave
#pragma unroll
    for (int r = 0; r < 8; ++r) {
      const int grow = qt + h * 8 + r;
      float v0 = s0.f[r], v1 = s1.f[r];
      if (causal) {
        if (kc + l16 > grow)      v0 = -1e30f;
        if (kc + 16 + l16 > grow) v1 = -1e30f;
      }
      float mx = fmaxf(v0, v1);
#pragma unroll
      for (int off = 1; off < 16; off <<= 1)
        mx = fmaxf(mx, __shfl_xor(mx, off, 32));
      const float mn = fmaxf(mrow[r], mx);
      const float al = __expf(mrow[r] - mn);
      const float p0 = __expf(v0 - mn);
      const float p1 = __expf(v1 - mn);
      float ps = p0 + p1;
#pragma unroll
      for (int off = 1; off < 16; off <<= 1) ps += __shfl_xor(ps, off, 32);
      lrow[r] = lrow[r] * al + ps;
      mrow[r] = mn;
      Pl[wv][(h * 8 + r) * 40 + l16]      = f2bf(p0);
      Pl[wv][(h * 8 + r) * 40 + 16 + l16] = f2bf(p1);
#pragma unroll
      for (int n = 0; n < 4; ++n) o[n].f[r] *= al;
    }
    // reload P in A-fragment layout (same wave; ordered via dscnt waits)
    FragA pa;
    const u16* prow = &Pl[wv][l16 * 40 + h * 8];
    pa.u4[0] = *(const uint4*)(prow);
    pa.u4[1] = *(const uint4*)(prow + 16);
#pragma unroll
    for (int n = 0; n < 4; ++n) {             // O dims in 4 tiles of 16
      FragA bv;
      const u16* vr = Vp + (size_t)(n * 16 + l16) * nk + kc + h * 8;
      bv.u4[0] = *(const uint4*)(vr);
      bv.u4[1] = *(const uint4*)(vr + 16);
      o[n].v = wmma_bf16(pa.v, bv.v, o[n].v);
    }
  }
  float inv[8];
#pragma unroll
  for (int r = 0; r < 8; ++r) inv[r] = 1.0f / lrow[r];
#pragma unroll
  for (int n = 0; n < 4; ++n)
#pragma unroll
    for (int r = 0; r < 8; ++r) {
      const int grow = qt + h * 8 + r;
      O[(size_t)grow * 1024 + head * 64 + n * 16 + l16] =
          f2bf(o[n].f[r] * inv[r]);
    }
}

// ---------------------------------------------------------------------------
// LayerNorm (row of 1024) -> bf16
// ---------------------------------------------------------------------------
__global__ __launch_bounds__(256) void ln_bf16_kernel(
    const float* __restrict__ x, const float* __restrict__ g,
    const float* __restrict__ b, u16* __restrict__ out)
{
  __shared__ float red[8], red2[8];
  const int row = blockIdx.x, tid = threadIdx.x;
  const int wv = tid >> 5, lane = tid & 31;
  const float* xr = x + (size_t)row * 1024;
  float v[4]; float s = 0.f;
#pragma unroll
  for (int i = 0; i < 4; ++i) { v[i] = xr[tid + i * 256]; s += v[i]; }
#pragma unroll
  for (int off = 1; off < 32; off <<= 1) s += __shfl_xor(s, off, 32);
  if (lane == 0) red[wv] = s;
  __syncthreads();
  float tot = 0.f;
#pragma unroll
  for (int j = 0; j < 8; ++j) tot += red[j];
  const float mu = tot * (1.0f / 1024.0f);
  float vs = 0.f;
#pragma unroll
  for (int i = 0; i < 4; ++i) { float d = v[i] - mu; vs += d * d; }
#pragma unroll
  for (int off = 1; off < 32; off <<= 1) vs += __shfl_xor(vs, off, 32);
  if (lane == 0) red2[wv] = vs;
  __syncthreads();
  float vt = 0.f;
#pragma unroll
  for (int j = 0; j < 8; ++j) vt += red2[j];
  const float rs = rsqrtf(vt * (1.0f / 1024.0f) + 1e-5f);
#pragma unroll
  for (int i = 0; i < 4; ++i) {
    const int c = tid + i * 256;
    out[(size_t)row * 1024 + c] = f2bf((v[i] - mu) * rs * g[c] + b[c]);
  }
}

// rotary on q,k,v (reference rotates all three), q scaled by 64^-0.5;
// packs Qh/Kh row-major per head, V dim-major for b128 fragment loads.
__global__ void rotary_pack_kernel(const float* __restrict__ qkv,
                                   const float* __restrict__ freqs,
                                   u16* __restrict__ Qh, u16* __restrict__ Kh,
                                   u16* __restrict__ Vt, int n)
{
  const int idx = blockIdx.x * blockDim.x + threadIdx.x;
  if (idx >= n * 16 * 32) return;
  const int pr = idx & 31;
  const int hd = (idx >> 5) & 15;
  const int row = idx >> 9;
  const int d0 = pr * 2;
  const float f = freqs[row * 64 + d0];     // freqs repeated per pair
  float sn, cs;
  sincosf(f, &sn, &cs);
  const float* base = qkv + (size_t)row * 3072 + hd * 64 + d0;
  {
    const float x1 = base[0], x2 = base[1];
    u16* q = Qh + ((size_t)hd * n + row) * 64 + d0;
    q[0] = f2bf((x1 * cs - x2 * sn) * 0.125f);
    q[1] = f2bf((x2 * cs + x1 * sn) * 0.125f);
  }
  {
    const float x1 = base[1024], x2 = base[1025];
    u16* k = Kh + ((size_t)hd * n + row) * 64 + d0;
    k[0] = f2bf(x1 * cs - x2 * sn);
    k[1] = f2bf(x2 * cs + x1 * sn);
  }
  {
    const float x1 = base[2048], x2 = base[2049];
    Vt[((size_t)hd * 64 + d0) * n + row]     = f2bf(x1 * cs - x2 * sn);
    Vt[((size_t)hd * 64 + d0 + 1) * n + row] = f2bf(x2 * cs + x1 * sn);
  }
}

__global__ void pack_q_kernel(const float* __restrict__ src,
                              u16* __restrict__ Qh, int n)
{
  const int idx = blockIdx.x * blockDim.x + threadIdx.x;
  if (idx >= n * 1024) return;
  const int row = idx >> 10, c = idx & 1023;
  const int hd = c >> 6, d = c & 63;
  Qh[((size_t)hd * n + row) * 64 + d] = f2bf(src[idx] * 0.125f);
}

__global__ void pack_kv_kernel(const float* __restrict__ src,
                               u16* __restrict__ Kh, u16* __restrict__ Vt,
                               int n)
{
  const int idx = blockIdx.x * blockDim.x + threadIdx.x;
  if (idx >= n * 1024) return;
  const int row = idx >> 10, c = idx & 1023;
  const int hd = c >> 6, d = c & 63;
  Kh[((size_t)hd * n + row) * 64 + d] = f2bf(src[(size_t)row * 2048 + c]);
  Vt[((size_t)hd * 64 + d) * n + row] = f2bf(src[(size_t)row * 2048 + 1024 + c]);
}

__global__ void geglu_kernel(const float* __restrict__ hf,
                             u16* __restrict__ out, int total)
{
  const int idx = blockIdx.x * blockDim.x + threadIdx.x;
  if (idx >= total) return;
  const int row = idx >> 12, c = idx & 4095;
  const float u = hf[(size_t)row * 8192 + c];
  const float g = hf[(size_t)row * 8192 + 4096 + c];
  const float ge = 0.5f * g * (1.0f + erff(g * 0.70710678118f));  // exact gelu
  out[idx] = f2bf(u * ge);
}

__global__ void cvt_bf16_kernel(const float* __restrict__ src,
                                u16* __restrict__ dst, int n)
{
  const int idx = blockIdx.x * blockDim.x + threadIdx.x;
  if (idx < n) dst[idx] = f2bf(src[idx]);
}

// Tiled transpose+convert: src fp32 [K][N] -> dst bf16 [N][K]
// (weights become B^T so GEMM B-tiles are contiguous 2D copies for the TDM).
__global__ __launch_bounds__(256) void cvt_transpose_kernel(
    const float* __restrict__ src, u16* __restrict__ dst, int K, int N)
{
  __shared__ float tile[32][33];
  const int k0 = blockIdx.y * 32, n0 = blockIdx.x * 32;
  const int tx = threadIdx.x & 31, ty = threadIdx.x >> 5;   // ty: 0..7
#pragma unroll
  for (int i = ty; i < 32; i += 8)
    tile[i][tx] = src[(size_t)(k0 + i) * N + n0 + tx];
  __syncthreads();
#pragma unroll
  for (int i = ty; i < 32; i += 8)
    dst[(size_t)(n0 + i) * K + k0 + tx] = f2bf(tile[tx][i]);
}

// ---------------------------------------------------------------------------
extern "C" void kernel_launch(void* const* d_in, const int* in_sizes, int n_in,
                              void* d_out, int out_size, void* d_ws, size_t ws_size,
                              hipStream_t stream)
{
  (void)in_sizes; (void)n_in; (void)out_size; (void)ws_size;
  const int N = 2048, Mc = 256, D = 1024, H = 16, L = 2;

  const float* x_in   = (const float*)d_in[0];
  const float* cond   = (const float*)d_in[2];
  const float* freqs  = (const float*)d_in[4];
  const float* ln1_g  = (const float*)d_in[5];
  const float* ln1_b  = (const float*)d_in[6];
  const float* qkv_w  = (const float*)d_in[7];
  const float* aout_w = (const float*)d_in[8];
  const float* aout_b = (const float*)d_in[9];
  const float* ls1    = (const float*)d_in[10];
  const float* ln2_g  = (const float*)d_in[11];
  const float* ln2_b  = (const float*)d_in[12];
  const float* q_w    = (const float*)d_in[13];
  const float* kv_w   = (const float*)d_in[14];
  const float* cout_w = (const float*)d_in[15];
  const float* cout_b = (const float*)d_in[16];
  const float* ls2    = (const float*)d_in[17];
  const float* ln3_g  = (const float*)d_in[18];
  const float* ln3_b  = (const float*)d_in[19];
  const float* ff_w1  = (const float*)d_in[20];
  const float* ff_b1  = (const float*)d_in[21];
  const float* ff_w2  = (const float*)d_in[22];
  const float* ff_b2  = (const float*)d_in[23];
  const float* ls3    = (const float*)d_in[24];

  char* ws = (char*)d_ws;
  u16*   wbf    = (u16*)(ws);                      // 41,943,040 B: bf16 W^T (per layer, reused)
  u16*   abf    = (u16*)(ws + 41943040ull);        //  4,194,304 B: LN output bf16
  u16*   condbf = (u16*)(ws + 46137344ull);        //    524,288 B: cond bf16
  u16*   Qh     = (u16*)(ws + 46661632ull);        //  4,194,304 B
  u16*   Kh     = (u16*)(ws + 50855936ull);        //  4,194,304 B
  u16*   Vt     = (u16*)(ws + 55050240ull);        //  4,194,304 B
  u16*   Obf    = (u16*)(ws + 59244544ull);        //  4,194,304 B: merged-head attn out
  u16*   gbf    = (u16*)(ws + 63438848ull);        // 16,777,216 B: geglu out bf16
  float* bigf   = (float*)(ws + 80216064ull);      // 67,108,864 B: fp32 GEMM scratch

  float* xbuf = (float*)d_out;                     // working residual stream
  hipMemcpyAsync(xbuf, x_in, (size_t)N * D * sizeof(float),
                 hipMemcpyDeviceToDevice, stream);

  { // cond -> bf16 (once)
    const int n = Mc * D;
    cvt_bf16_kernel<<<(n + 255) / 256, 256, 0, stream>>>(cond, condbf, n);
  }

  // per-layer transposed bf16 weight sub-offsets (elements)
  u16* w_qkv = wbf;                  // [3072][1024]
  u16* w_ao  = wbf + 3145728ull;     // [1024][1024]
  u16* w_q   = wbf + 4194304ull;     // [1024][1024]
  u16* w_kv  = wbf + 5242880ull;     // [2048][1024]
  u16* w_co  = wbf + 7340032ull;     // [1024][1024]
  u16* w_f1  = wbf + 8388608ull;     // [8192][1024]
  u16* w_f2  = wbf + 16777216ull;    // [1024][4096]

  auto cvtT = [&](const float* s, u16* d, int K, int Nn) {
    cvt_transpose_kernel<<<dim3(Nn / 32, K / 32), 256, 0, stream>>>(s, d, K, Nn);
  };
  auto gemm = [&](const u16* A, const u16* B, const float* bias,
                  const float* res, const float* lsv, float* C,
                  int Mm, int Nn, int Kk) {
    dim3 g(Nn / 128, Mm / 128);
    gemm_bf16_kernel<<<g, 256, 40960, stream>>>(A, B, bias, res, lsv, C, Mm, Nn, Kk);
  };

  for (int l = 0; l < L; ++l) {
    cvtT(qkv_w  + (size_t)l * 3145728, w_qkv, D, 3 * D);
    cvtT(aout_w + (size_t)l * 1048576, w_ao,  D, D);
    cvtT(q_w    + (size_t)l * 1048576, w_q,   D, D);
    cvtT(kv_w   + (size_t)l * 2097152, w_kv,  D, 2 * D);
    cvtT(cout_w + (size_t)l * 1048576, w_co,  D, D);
    cvtT(ff_w1  + (size_t)l * 8388608, w_f1,  D, 8192);
    cvtT(ff_w2  + (size_t)l * 4194304, w_f2,  4096, D);

    // ---- self attention (causal) ----
    ln_bf16_kernel<<<N, 256, 0, stream>>>(xbuf, ln1_g + l * D, ln1_b + l * D, abf);
    gemm(abf, w_qkv, nullptr, nullptr, nullptr, bigf, N, 3 * D, D);
    {
      const int n = N * H * 32;
      rotary_pack_kernel<<<(n + 255) / 256, 256, 0, stream>>>(bigf, freqs, Qh, Kh, Vt, N);
    }
    attn_kernel<<<dim3(N / 64, H), 256, 0, stream>>>(Qh, Kh, Vt, Obf, N, N, 1);
    gemm(Obf, w_ao, aout_b + l * D, xbuf, ls1 + l * D, xbuf, N, D, D);

    // ---- cross attention ----
    ln_bf16_kernel<<<N, 256, 0, stream>>>(xbuf, ln2_g + l * D, ln2_b + l * D, abf);
    gemm(abf, w_q, nullptr, nullptr, nullptr, bigf, N, D, D);
    {
      const int n = N * D;
      pack_q_kernel<<<(n + 255) / 256, 256, 0, stream>>>(bigf, Qh, N);
    }
    gemm(condbf, w_kv, nullptr, nullptr, nullptr, bigf, Mc, 2 * D, D);
    {
      const int n = Mc * D;
      pack_kv_kernel<<<(n + 255) / 256, 256, 0, stream>>>(bigf, Kh, Vt, Mc);
    }
    attn_kernel<<<dim3(N / 64, H), 256, 0, stream>>>(Qh, Kh, Vt, Obf, N, Mc, 0);
    gemm(Obf, w_co, cout_b + l * D, xbuf, ls2 + l * D, xbuf, N, D, D);

    // ---- GEGLU feed-forward ----
    ln_bf16_kernel<<<N, 256, 0, stream>>>(xbuf, ln3_g + l * D, ln3_b + l * D, abf);
    gemm(abf, w_f1, ff_b1 + (size_t)l * 8192, nullptr, nullptr, bigf, N, 8192, D);
    {
      const int n = N * 4096;
      geglu_kernel<<<(n + 255) / 256, 256, 0, stream>>>(bigf, gbf, n);
    }
    gemm(gbf, w_f2, ff_b2 + l * D, xbuf, ls3 + l * D, xbuf, N, D, 4096);
  }
}
